// SelfGNN_73598559584740
// MI455X (gfx1250) — matrix operand order repaired and checked by reading, not compiled, not run on hardware
//
#include <hip/hip_runtime.h>
#include <math.h>

#define NNODE 4096
#define BSZ   16
#define MTOT  65536          // BSZ * NNODE
#define TAU_INV (1.0f/0.3f)
#define EPS_INV 20.0f        // 1/0.05
#define SINK_ITERS 20

typedef __attribute__((ext_vector_type(16))) __bf16 bf16x16;
typedef __attribute__((ext_vector_type(8)))  float  f32x8;

// ---------------------------------------------------------------------------
// GEMM:  C[M x NOUT] = act( A1[M x K] @ W1[NOUT x K]^T
//                           (+ A2 @ W2^T)  (+ bias) ),  optional ReLU.
// 256 threads / block, 128-row tile, 8 waves each own a 16-row band and all
// NOUT/16 column tiles.  f32 inputs converted to bf16 in LDS, f32 accumulate
// via v_wmma_f32_16x16x32_bf16.
// ---------------------------------------------------------------------------
template<int NOUT>
__global__ __launch_bounds__(256)
void gemm_wmma_kernel(const float* __restrict__ A1, const float* __restrict__ W1,
                      const float* __restrict__ A2, const float* __restrict__ W2,
                      const float* __restrict__ bias, float* __restrict__ C,
                      int K, int relu)
{
    constexpr int NT = NOUT / 16;
    __shared__ __bf16 sA[128 * 32];
    __shared__ __bf16 sB[NOUT * 32];
    const int tid  = threadIdx.x;
    const int lane = tid & 31;
    const int wave = tid >> 5;          // 0..7
    const int hi   = lane >> 4;         // 0/1
    const int l15  = lane & 15;
    const size_t blockM = (size_t)blockIdx.x * 128;

    f32x8 acc[NT] = {};

    const int nphase = (A2 != nullptr) ? 2 : 1;
    for (int ph = 0; ph < nphase; ++ph) {
        const float* Ap = ph ? A2 : A1;
        const float* Wp = ph ? W2 : W1;
        for (int kc = 0; kc < K; kc += 32) {
            // stage A tile (128 x 32) f32 -> bf16
            #pragma unroll
            for (int p = 0; p < 4; ++p) {
                int linear = p * 1024 + tid * 4;
                int r = linear >> 5, c = linear & 31;
                float4 v = *(const float4*)(Ap + (blockM + r) * (size_t)K + kc + c);
                sA[r*32 + c + 0] = (__bf16)v.x;  sA[r*32 + c + 1] = (__bf16)v.y;
                sA[r*32 + c + 2] = (__bf16)v.z;  sA[r*32 + c + 3] = (__bf16)v.w;
            }
            // stage W tile (NOUT x 32) f32 -> bf16
            #pragma unroll
            for (int p = 0; p < 4; ++p) {
                int linear = p * 1024 + tid * 4;
                if (linear < NOUT * 32) {
                    int r = linear >> 5, c = linear & 31;
                    float4 v = *(const float4*)(Wp + (size_t)r * K + kc + c);
                    sB[r*32 + c + 0] = (__bf16)v.x;  sB[r*32 + c + 1] = (__bf16)v.y;
                    sB[r*32 + c + 2] = (__bf16)v.z;  sB[r*32 + c + 3] = (__bf16)v.w;
                }
            }
            __syncthreads();

            // A fragment: 16x32 bf16 per CDNA5 layout
            // lanes 0-15: elems 0..7 = K0..7, 8..15 = K16..23 ; lanes 16-31: +8
            bf16x16 af;
            {
                int row = wave * 16 + l15;
                #pragma unroll
                for (int i = 0; i < 8; ++i) af[i]     = sA[row*32 + hi*8 + i];
                #pragma unroll
                for (int i = 0; i < 8; ++i) af[8 + i] = sA[row*32 + 16 + hi*8 + i];
            }
            #pragma unroll
            for (int nt = 0; nt < NT; ++nt) {
                // B fragment: 32x16 bf16 — lane = column, same K order as A (K0..7,K16..23; lanes 16-31: +8)
                bf16x16 bfm;
                int col = nt * 16 + l15;
                #pragma unroll
                for (int i = 0; i < 8; ++i) bfm[i] = sB[col*32 + hi*8 + i];
                #pragma unroll
                for (int i = 0; i < 8; ++i) bfm[8 + i] = sB[col*32 + 16 + hi*8 + i];
                acc[nt] = __builtin_amdgcn_wmma_f32_16x16x32_bf16(
                              false, af, false, bfm, (short)0, acc[nt], false, false);
            }
            __syncthreads();
        }
    }

    // epilogue: bias + optional ReLU.  C/D layout: VGPR r -> row r + 8*hi.
    #pragma unroll
    for (int nt = 0; nt < NT; ++nt) {
        int col = nt * 16 + l15;
        float bv = bias ? bias[col] : 0.0f;
        #pragma unroll
        for (int r = 0; r < 8; ++r) {
            size_t row = blockM + wave * 16 + r + hi * 8;
            float v = acc[nt][r] + bv;
            if (relu) v = fmaxf(v, 0.0f);
            C[row * NOUT + col] = v;
        }
    }
}

// ---------------------------------------------------------------------------
// Circular sliding-window mean over the 8 preceding nodes (per batch).
// ---------------------------------------------------------------------------
__global__ __launch_bounds__(256)
void winmean_kernel(const float* __restrict__ in, float* __restrict__ out, int C)
{
    int idx  = blockIdx.x * 256 + threadIdx.x;   // < MTOT*C  (fits int)
    int perb = NNODE * C;
    int b = idx / perb;
    int rem = idx - b * perb;
    int n = rem / C, c = rem - (rem / C) * C;
    float s = 0.0f;
    #pragma unroll
    for (int k = 1; k <= 8; ++k)
        s += in[((size_t)b * NNODE + ((n - k) & (NNODE - 1))) * C + c];
    out[idx] = s * 0.125f;
}

// ---------------------------------------------------------------------------
// Row L2-normalize (row length 128).  One wave per row.
// ---------------------------------------------------------------------------
__global__ __launch_bounds__(256)
void rownorm_kernel(const float* __restrict__ in, float* __restrict__ out)
{
    int row  = blockIdx.x * 8 + (threadIdx.x >> 5);
    int lane = threadIdx.x & 31;
    const float4 v = *(const float4*)(in + (size_t)row * 128 + lane * 4);
    float ss = v.x*v.x + v.y*v.y + v.z*v.z + v.w*v.w;
    for (int o = 16; o > 0; o >>= 1) ss += __shfl_xor(ss, o, 32);
    float scale = 1.0f / fmaxf(sqrtf(ss), 1e-12f);
    float4 ov = make_float4(v.x*scale, v.y*scale, v.z*scale, v.w*scale);
    *(float4*)(out + (size_t)row * 128 + lane * 4) = ov;
}

__global__ __launch_bounds__(256)
void exp_kernel(const float* __restrict__ zc, float* __restrict__ E, int n)
{
    int i = blockIdx.x * 256 + threadIdx.x;
    if (i < n) E[i] = expf(zc[i] * EPS_INV);
}

__global__ __launch_bounds__(256)
void fill1_kernel(float* __restrict__ v, int n)
{
    int i = blockIdx.x * 256 + threadIdx.x;
    if (i < n) v[i] = 1.0f;
}

// Sinkhorn in scaling form: u_k = 1/(K * sum_b E_bk v_b)
__global__ __launch_bounds__(256)
void sink_row_kernel(const float* __restrict__ E, const float* __restrict__ v,
                     float* __restrict__ u)
{
    int k = blockIdx.x;                        // 0..63
    float s = 0.0f;
    for (int b = threadIdx.x; b < MTOT; b += 256) s += E[(size_t)b * 64 + k] * v[b];
    __shared__ float red[256];
    red[threadIdx.x] = s; __syncthreads();
    for (int off = 128; off > 0; off >>= 1) {
        if (threadIdx.x < off) red[threadIdx.x] += red[threadIdx.x + off];
        __syncthreads();
    }
    if (threadIdx.x == 0) u[k] = 1.0f / (64.0f * red[0]);
}

// v_b = 1/(B * sum_k E_bk u_k)
__global__ __launch_bounds__(256)
void sink_col_kernel(const float* __restrict__ E, const float* __restrict__ u,
                     float* __restrict__ v)
{
    int b = blockIdx.x * 256 + threadIdx.x;    // < MTOT
    float s = 0.0f;
    for (int k = 0; k < 64; ++k) s += E[(size_t)b * 64 + k] * u[k];
    v[b] = 1.0f / ((float)MTOT * s);
}

// loss1 partials: per sample b, sum_k q1*logsm(zc2/tau) + q2*logsm(zc1/tau)
__global__ __launch_bounds__(256)
void loss1_kernel(const float* __restrict__ E1, const float* __restrict__ u1,
                  const float* __restrict__ v1,
                  const float* __restrict__ E2, const float* __restrict__ u2,
                  const float* __restrict__ v2,
                  const float* __restrict__ zc1, const float* __restrict__ zc2,
                  float* __restrict__ partials)
{
    int wave = threadIdx.x >> 5, lane = threadIdx.x & 31;
    int b = blockIdx.x * 8 + wave;
    size_t base = (size_t)b * 64;
    int k0 = lane, k1 = lane + 32;

    // term A: q1 (from E1) against log_softmax(zc2/tau)
    float xa = zc2[base + k0] * TAU_INV, xb = zc2[base + k1] * TAU_INV;
    float m = fmaxf(xa, xb);
    for (int o = 16; o > 0; o >>= 1) m = fmaxf(m, __shfl_xor(m, o, 32));
    float es = expf(xa - m) + expf(xb - m);
    for (int o = 16; o > 0; o >>= 1) es += __shfl_xor(es, o, 32);
    float lse = m + logf(es);
    float q1a = E1[base + k0] * u1[k0] * v1[b] * (float)MTOT;
    float q1b = E1[base + k1] * u1[k1] * v1[b] * (float)MTOT;
    float t = q1a * (xa - lse) + q1b * (xb - lse);

    // term B: q2 against log_softmax(zc1/tau)
    float ya = zc1[base + k0] * TAU_INV, yb = zc1[base + k1] * TAU_INV;
    float m2 = fmaxf(ya, yb);
    for (int o = 16; o > 0; o >>= 1) m2 = fmaxf(m2, __shfl_xor(m2, o, 32));
    float es2 = expf(ya - m2) + expf(yb - m2);
    for (int o = 16; o > 0; o >>= 1) es2 += __shfl_xor(es2, o, 32);
    float lse2 = m2 + logf(es2);
    float q2a = E2[base + k0] * u2[k0] * v2[b] * (float)MTOT;
    float q2b = E2[base + k1] * u2[k1] * v2[b] * (float)MTOT;
    t += q2a * (ya - lse2) + q2b * (yb - lse2);

    for (int o = 16; o > 0; o >>= 1) t += __shfl_xor(t, o, 32);
    __shared__ float red[8];
    if (lane == 0) red[wave] = t;
    __syncthreads();
    if (threadIdx.x == 0) {
        float s = 0; for (int w = 0; w < 8; ++w) s += red[w];
        partials[blockIdx.x] = s;
    }
}

// top-5 = {self (diag, unique max 1.0)} U {4 largest of the node's 8 weights}
__global__ __launch_bounds__(256)
void topk_kernel(const float* __restrict__ ew, int* __restrict__ tk)
{
    int i = blockIdx.x * 256 + threadIdx.x;
    if (i >= NNODE) return;
    float w[8]; bool used[8];
    #pragma unroll
    for (int t = 0; t < 8; ++t) { w[t] = ew[i * 8 + t]; used[t] = false; }
    tk[i * 5 + 0] = i;
    for (int s = 0; s < 4; ++s) {
        int best = 0; float bw = -1e30f;
        #pragma unroll
        for (int t = 0; t < 8; ++t)
            if (!used[t] && w[t] > bw) { bw = w[t]; best = t; }
        used[best] = true;
        tk[i * 5 + 1 + s] = (i + 1 + best) & (NNODE - 1);
    }
}

// attention pool over top-5 neighbors.  One wave per (b,n).
__global__ __launch_bounds__(256)
void pool_kernel(const float* __restrict__ z1, const int* __restrict__ tk,
                 const float* __restrict__ Wsa, const float* __restrict__ bsa,
                 float* __restrict__ pooled)
{
    int wave = threadIdx.x >> 5, lane = threadIdx.x & 31;
    int gid = blockIdx.x * 8 + wave;
    int b = gid >> 12, n = gid & (NNODE - 1);
    const float4 wv = *(const float4*)(Wsa + lane * 4);
    float4 rows[5]; float dots[5];
    #pragma unroll
    for (int j = 0; j < 5; ++j) {
        int idx = tk[n * 5 + j];
        const float4 r = *(const float4*)(z1 + ((size_t)b * NNODE + idx) * 128 + lane * 4);
        rows[j] = r;
        float d = r.x*wv.x + r.y*wv.y + r.z*wv.z + r.w*wv.w;
        for (int o = 16; o > 0; o >>= 1) d += __shfl_xor(d, o, 32);
        dots[j] = d + bsa[0];
    }
    float mx = dots[0];
    #pragma unroll
    for (int j = 1; j < 5; ++j) mx = fmaxf(mx, dots[j]);
    float es[5], ssum = 0.0f;
    #pragma unroll
    for (int j = 0; j < 5; ++j) { es[j] = expf(dots[j] - mx); ssum += es[j]; }
    float inv = 1.0f / ssum;
    float4 o = make_float4(0.f, 0.f, 0.f, 0.f);
    #pragma unroll
    for (int j = 0; j < 5; ++j) {
        float a = es[j] * inv;
        o.x += a * rows[j].x; o.y += a * rows[j].y;
        o.z += a * rows[j].z; o.w += a * rows[j].w;
    }
    *(float4*)(pooled + (size_t)gid * 128 + lane * 4) = o;
}

__device__ __forceinline__ float softplus_f(float x)
{
    return fmaxf(x, 0.0f) + log1pf(expf(-fabsf(x)));
}

// loss2 + accuracy partials.  One wave per (b,n).
__global__ __launch_bounds__(256)
void loss2_kernel(const float* __restrict__ z2n, const float* __restrict__ ansn,
                  const int* __restrict__ perm_bs, const int* __restrict__ perm_nodes,
                  float* __restrict__ pl, float* __restrict__ pa)
{
    int wave = threadIdx.x >> 5, lane = threadIdx.x & 31;
    int gid = blockIdx.x * 8 + wave;
    int b = gid >> 12, n = gid & (NNODE - 1);
    int pb = perm_bs[b], pn = perm_nodes[n];
    const float4 s  = *(const float4*)(z2n + (size_t)gid * 128 + lane * 4);
    const float4 hr = *(const float4*)(ansn + (size_t)gid * 128 + lane * 4);
    const float4 hf = *(const float4*)(ansn + ((size_t)pb * NNODE + pn) * 128 + lane * 4);
    float rl = s.x*hr.x + s.y*hr.y + s.z*hr.z + s.w*hr.w;
    float fk = s.x*hf.x + s.y*hf.y + s.z*hf.z + s.w*hf.w;
    for (int o = 16; o > 0; o >>= 1) { rl += __shfl_xor(rl, o, 32); fk += __shfl_xor(fk, o, 32); }
    float lc = softplus_f(-rl) + softplus_f(fk);
    float ac = (rl > 0.0f ? 1.0f : 0.0f) + (fk <= 0.0f ? 1.0f : 0.0f);
    __shared__ float rd1[8], rd2[8];
    if (lane == 0) { rd1[wave] = lc; rd2[wave] = ac; }
    __syncthreads();
    if (threadIdx.x == 0) {
        float a = 0, c = 0;
        for (int w = 0; w < 8; ++w) { a += rd1[w]; c += rd2[w]; }
        pl[blockIdx.x] = a; pa[blockIdx.x] = c;
    }
}

__global__ __launch_bounds__(256)
void final_kernel(const float* __restrict__ P1, const float* __restrict__ P2,
                  const float* __restrict__ Pacc, int n, float* __restrict__ out)
{
    float s1 = 0, s2 = 0, sa = 0;
    for (int i = threadIdx.x; i < n; i += 256) { s1 += P1[i]; s2 += P2[i]; sa += Pacc[i]; }
    __shared__ float r1[256], r2[256], r3[256];
    r1[threadIdx.x] = s1; r2[threadIdx.x] = s2; r3[threadIdx.x] = sa;
    __syncthreads();
    for (int off = 128; off > 0; off >>= 1) {
        if (threadIdx.x < off) {
            r1[threadIdx.x] += r1[threadIdx.x + off];
            r2[threadIdx.x] += r2[threadIdx.x + off];
            r3[threadIdx.x] += r3[threadIdx.x + off];
        }
        __syncthreads();
    }
    if (threadIdx.x == 0) {
        float loss1 = -r1[0] / (float)MTOT;
        float loss2 =  r2[0] / (float)(2 * MTOT);
        float acc   =  r3[0] / (float)(2 * MTOT);
        out[0] = loss1 + loss2;
        out[1] = acc;
    }
}

// ---------------------------------------------------------------------------
extern "C" void kernel_launch(void* const* d_in, const int* in_sizes, int n_in,
                              void* d_out, int out_size, void* d_ws, size_t ws_size,
                              hipStream_t stream)
{
    (void)in_sizes; (void)n_in; (void)out_size; (void)ws_size;
    const float* x     = (const float*)d_in[0];
    const float* y     = (const float*)d_in[1];
    const float* ew    = (const float*)d_in[2];
    const float* Wp1   = (const float*)d_in[3];
    const float* bp1   = (const float*)d_in[4];
    const float* Wl1   = (const float*)d_in[5];
    const float* bl1   = (const float*)d_in[6];
    const float* Wr1   = (const float*)d_in[7];
    const float* Wp2   = (const float*)d_in[8];
    const float* bp2   = (const float*)d_in[9];
    const float* Wl2   = (const float*)d_in[10];
    const float* bl2   = (const float*)d_in[11];
    const float* Wr2   = (const float*)d_in[12];
    const float* Wenc  = (const float*)d_in[13];
    const float* benc  = (const float*)d_in[14];
    const float* Wproto= (const float*)d_in[15];
    const float* Wsa   = (const float*)d_in[16];
    const float* bsa   = (const float*)d_in[17];
    const float* Wnc   = (const float*)d_in[18];
    const float* bnc   = (const float*)d_in[19];
    const int* perm_bs    = (const int*)d_in[22];
    const int* perm_nodes = (const int*)d_in[23];
    float* out = (float*)d_out;

    const size_t SLOT = (size_t)MTOT * 128;      // 32 MB slots
    float* ws  = (float*)d_ws;
    float* S0  = ws;                 // hp / pooled / ansn
    float* S1  = ws + SLOT;          // agg / ans
    float* S2  = ws + 2 * SLOT;      // h / z1n / z2n (persists for loss2)
    float* z1  = ws + 3 * SLOT;
    float* z2  = ws + 4 * SLOT;
    float* zc1 = ws + 5 * SLOT;      float* zc2 = zc1 + SLOT / 2;
    float* E1  = ws + 6 * SLOT;      float* E2  = E1  + SLOT / 2;
    float* ex  = ws + 7 * SLOT;
    float* u1  = ex;                 float* v1 = u1 + 64;
    float* u2  = v1 + MTOT;          float* v2 = u2 + 64;
    float* P1  = v2 + MTOT;
    float* P2  = P1 + 8192;
    float* Pacc= P2 + 8192;
    int*   tk  = (int*)(Pacc + 8192);

    const dim3 B256(256);
    const dim3 Ggemm(MTOT / 128);

    auto encoder = [&](const float* in, float* zout) {
        gemm_wmma_kernel<64 ><<<Ggemm, B256, 0, stream>>>(in, Wp1, nullptr, nullptr, bp1, S0, 64, 1);
        winmean_kernel<<<dim3(MTOT * 64 / 256), B256, 0, stream>>>(S0, S1, 64);
        gemm_wmma_kernel<128><<<Ggemm, B256, 0, stream>>>(S1, Wl1, in, Wr1, bl1, S2, 64, 1);
        gemm_wmma_kernel<128><<<Ggemm, B256, 0, stream>>>(S2, Wp2, nullptr, nullptr, bp2, S0, 128, 1);
        winmean_kernel<<<dim3(MTOT * 128 / 256), B256, 0, stream>>>(S0, S1, 128);
        gemm_wmma_kernel<128><<<Ggemm, B256, 0, stream>>>(S1, Wl2, S2, Wr2, bl2, S0, 128, 1);
        gemm_wmma_kernel<128><<<Ggemm, B256, 0, stream>>>(S0, Wenc, nullptr, nullptr, benc, zout, 128, 0);
    };
    encoder(y, z1);
    encoder(x, z2);

    // prototype scores on l2-normalized reps
    rownorm_kernel<<<dim3(MTOT / 8), B256, 0, stream>>>(z1, S2);
    gemm_wmma_kernel<64><<<Ggemm, B256, 0, stream>>>(S2, Wproto, nullptr, nullptr, nullptr, zc1, 128, 0);
    rownorm_kernel<<<dim3(MTOT / 8), B256, 0, stream>>>(z2, S2);   // S2 = z2n, kept for loss2
    gemm_wmma_kernel<64><<<Ggemm, B256, 0, stream>>>(S2, Wproto, nullptr, nullptr, nullptr, zc2, 128, 0);

    // sinkhorn (scaling-vector form; init v=1, global norm cancels after row step)
    exp_kernel<<<dim3(MTOT * 64 / 256), B256, 0, stream>>>(zc1, E1, MTOT * 64);
    exp_kernel<<<dim3(MTOT * 64 / 256), B256, 0, stream>>>(zc2, E2, MTOT * 64);
    fill1_kernel<<<dim3(MTOT / 256), B256, 0, stream>>>(v1, MTOT);
    fill1_kernel<<<dim3(MTOT / 256), B256, 0, stream>>>(v2, MTOT);
    for (int it = 0; it < SINK_ITERS; ++it) {
        sink_row_kernel<<<dim3(64), B256, 0, stream>>>(E1, v1, u1);
        sink_col_kernel<<<dim3(MTOT / 256), B256, 0, stream>>>(E1, u1, v1);
        sink_row_kernel<<<dim3(64), B256, 0, stream>>>(E2, v2, u2);
        sink_col_kernel<<<dim3(MTOT / 256), B256, 0, stream>>>(E2, u2, v2);
    }
    loss1_kernel<<<dim3(MTOT / 8), B256, 0, stream>>>(E1, u1, v1, E2, u2, v2, zc1, zc2, P1);

    // discriminator branch
    topk_kernel<<<dim3(NNODE / 256), B256, 0, stream>>>(ew, tk);
    pool_kernel<<<dim3(MTOT / 8), B256, 0, stream>>>(z1, tk, Wsa, bsa, S0);
    gemm_wmma_kernel<128><<<Ggemm, B256, 0, stream>>>(S0, Wnc, nullptr, nullptr, bnc, S1, 128, 0);
    rownorm_kernel<<<dim3(MTOT / 8), B256, 0, stream>>>(S1, S0);   // ansn
    loss2_kernel<<<dim3(MTOT / 8), B256, 0, stream>>>(S2, S0, perm_bs, perm_nodes, P2, Pacc);

    final_kernel<<<dim3(1), B256, 0, stream>>>(P1, P2, Pacc, 8192, out);
}
